// GatedResidualAttention_13889924235852
// MI455X (gfx1250) — compile-verified, matrix operand order checked by reading, not run on hardware
//
#include <hip/hip_runtime.h>
#include <hip/hip_bf16.h>
#include <math.h>
#include <stdint.h>

namespace {
constexpr int kB = 2, kT = 2048, kH = 16, kD = 64;
constexpr int kHID = kH * kD;   // 1024
constexpr int kBT  = kB * kT;   // 4096
constexpr int kBH  = kB * kH;   // 32
constexpr float kEps = 1e-6f;
}

typedef __bf16 bf16_t;
typedef __attribute__((ext_vector_type(16))) __bf16 v16bf;
typedef __attribute__((ext_vector_type(8)))  __bf16 v8bf;
typedef __attribute__((ext_vector_type(8)))  float  v8f;

// Load a 16x32 (bf16) WMMA A/B fragment slice for this lane.
// CDNA5 16-bit fragment layout (ISA 7.12.2): lane half lh=(lane>>4) holds
// K = lh*8 + {0..7} (elements 0..7) and K = 16 + lh*8 + {0..7} (elements 8..15):
// two contiguous 16-byte runs when the source is K-contiguous.
__device__ __forceinline__ v16bf load_frag_k32(const bf16_t* base, int lh) {
  v8bf lo = *reinterpret_cast<const v8bf*>(base + lh * 8);
  v8bf hi = *reinterpret_cast<const v8bf*>(base + 16 + lh * 8);
  v16bf r;
#pragma unroll
  for (int i = 0; i < 8; ++i) { r[i] = lo[i]; r[i + 8] = hi[i]; }
  return r;
}

__device__ __forceinline__ v8f wmma_bf16(v16bf a, v16bf b, v8f c) {
  return __builtin_amdgcn_wmma_f32_16x16x32_bf16(false, a, false, b, (short)0, c,
                                                 false, false);
}

// Intra-wave LDS producer/consumer fence (DS ops of one wave complete in order;
// this stops compiler reordering and drains DScnt).
__device__ __forceinline__ void wave_lds_fence() {
  asm volatile("s_wait_dscnt 0" ::: "memory");
}

// Async DMA of 16B from global memory into LDS (per lane). Tracked by ASYNCcnt.
__device__ __forceinline__ void async_load_b128(uint32_t lds_off, const void* gptr) {
  asm volatile("global_load_async_to_lds_b128 %0, %1, off"
               :: "v"(lds_off), "v"(gptr) : "memory");
}

// ---------------------------------------------------------------- converts --
__global__ void k_f32_to_bf16(const float* __restrict__ in, bf16_t* __restrict__ out,
                              int n) {
  int i = blockIdx.x * blockDim.x + threadIdx.x;
  if (i < n) out[i] = (bf16_t)in[i];
}

// out[n][k] = (bf16) in[k][n]  (store W^T so B fragments are K-contiguous)
__global__ void k_transpose_bf16(const float* __restrict__ in, bf16_t* __restrict__ out) {
  int i = blockIdx.x * blockDim.x + threadIdx.x; // kHID*kHID threads
  int k = i >> 10, n = i & (kHID - 1);
  out[(size_t)n * kHID + k] = (bf16_t)in[i];
}

// ------------------------------------------------- s_prev row mean/rsigma ---
__global__ void k_row_stats(const float* __restrict__ s_prev, float* __restrict__ mu,
                            float* __restrict__ rs) {
  const int row  = blockIdx.x * 8 + (threadIdx.x >> 5); // one wave per row
  const int lane = threadIdx.x & 31;
  const float* p = s_prev + (size_t)row * kT;
  float s = 0.f, q = 0.f;
  for (int j = lane; j < kT; j += 32) { float v = p[j]; s += v; q += v * v; }
#pragma unroll
  for (int off = 16; off > 0; off >>= 1) {
    s += __shfl_xor(s, off, 32);
    q += __shfl_xor(q, off, 32);
  }
  if (lane == 0) {
    float m   = s * (1.f / kT);
    float var = q * (1.f / kT) - m * m;
    mu[row] = m;
    rs[row] = rsqrtf(var + kEps);
  }
}

// ---------------------------------------------------------------- GEMM ------
// C = A(kBT x kHID bf16 row-major) * W, with Bt = W^T (kHID x kHID row-major).
// Wave computes a 16(M) x 64(N) strip; k-loop is software-pipelined so the
// loads of iteration k+1 overlap the four WMMAs of iteration k.
// mode 0: store bf16 (b,h,t,d) | mode 1: store bf16 (b,h,d,t) | mode 2: f32 MxN
__global__ void k_gemm_bf16(const bf16_t* __restrict__ A, const bf16_t* __restrict__ Bt,
                            void* __restrict__ outp, int mode) {
  const int lane = threadIdx.x & 31;
  const int wv   = threadIdx.x >> 5;
  const int lh = lane >> 4, ln = lane & 15;
  const int mtile = blockIdx.x * 64 + wv * 16;
  const int nbase = blockIdx.y * 64;

  v8f c[4];
#pragma unroll
  for (int j = 0; j < 4; ++j)
#pragma unroll
    for (int i = 0; i < 8; ++i) c[j][i] = 0.f;

  const bf16_t* arow = A + (size_t)(mtile + ln) * kHID;
  const bf16_t* b0 = Bt + (size_t)(nbase + 0 * 16 + ln) * kHID;
  const bf16_t* b1 = Bt + (size_t)(nbase + 1 * 16 + ln) * kHID;
  const bf16_t* b2 = Bt + (size_t)(nbase + 2 * 16 + ln) * kHID;
  const bf16_t* b3 = Bt + (size_t)(nbase + 3 * 16 + ln) * kHID;

  v16bf ac  = load_frag_k32(arow, lh);
  v16bf bc0 = load_frag_k32(b0, lh);
  v16bf bc1 = load_frag_k32(b1, lh);
  v16bf bc2 = load_frag_k32(b2, lh);
  v16bf bc3 = load_frag_k32(b3, lh);

#pragma unroll 2
  for (int kb = 0; kb < kHID - 32; kb += 32) {
    // prefetch next k-slice into fresh registers
    v16bf an  = load_frag_k32(arow + kb + 32, lh);
    v16bf bn0 = load_frag_k32(b0 + kb + 32, lh);
    v16bf bn1 = load_frag_k32(b1 + kb + 32, lh);
    v16bf bn2 = load_frag_k32(b2 + kb + 32, lh);
    v16bf bn3 = load_frag_k32(b3 + kb + 32, lh);
    c[0] = wmma_bf16(ac, bc0, c[0]);
    c[1] = wmma_bf16(ac, bc1, c[1]);
    c[2] = wmma_bf16(ac, bc2, c[2]);
    c[3] = wmma_bf16(ac, bc3, c[3]);
    ac = an; bc0 = bn0; bc1 = bn1; bc2 = bn2; bc3 = bn3;
  }
  c[0] = wmma_bf16(ac, bc0, c[0]);
  c[1] = wmma_bf16(ac, bc1, c[1]);
  c[2] = wmma_bf16(ac, bc2, c[2]);
  c[3] = wmma_bf16(ac, bc3, c[3]);

  const int mrow0 = mtile + 8 * lh; // C layout: VGPR r -> M = mrow0 + r, N = ln
#pragma unroll
  for (int j = 0; j < 4; ++j) {
    const int n = nbase + j * 16 + ln;
    const int h = n >> 6, d = n & 63;
#pragma unroll
    for (int r = 0; r < 8; ++r) {
      const int m = mrow0 + r;
      const int bb = m >> 11, t = m & (kT - 1);
      float v = c[j][r];
      if (mode == 0) {
        ((bf16_t*)outp)[((size_t)(bb * kH + h) * kT + t) * kD + d] = (bf16_t)v;
      } else if (mode == 1) {
        ((bf16_t*)outp)[((size_t)(bb * kH + h) * kD + d) * kT + t] = (bf16_t)v;
      } else {
        ((float*)outp)[(size_t)m * kHID + n] = v;
      }
    }
  }
}

// ------------------------------------------------ fused flash attention ----
// 8 waves/block, all sharing one (b,h); each wave owns 16 query rows.
// K/V tiles (32 keys) are DMA'd into double-buffered LDS with
// global_load_async_to_lds_b128 and consumed by all 8 waves; online softmax;
// `raw` (with gated residual) written on the fly.
__global__ void k_attn(const bf16_t* __restrict__ Qb, const bf16_t* __restrict__ Kb,
                       const bf16_t* __restrict__ Vt, const float* __restrict__ s_prev,
                       const float* __restrict__ mu, const float* __restrict__ rs,
                       const float* __restrict__ alp, const float* __restrict__ gmm,
                       const float* __restrict__ bet,
                       float* __restrict__ raw, bf16_t* __restrict__ ctx) {
  __shared__ __align__(16) bf16_t ldsK[2][32 * 64];   // [buf][key][d]
  __shared__ __align__(16) bf16_t ldsV[2][64 * 32];   // [buf][d][key]
  __shared__ __align__(16) float  ldsS[8][16 * 32];
  __shared__ __align__(16) bf16_t ldsP[8][16 * 32];
  __shared__ float ldsScale[8][16];
  __shared__ float ldsL[8][16];

  const int tid  = threadIdx.x;
  const int lane = tid & 31;
  const int wv   = tid >> 5;
  const int lh = lane >> 4, ln = lane & 15;
  const int task  = blockIdx.x * 8 + wv;
  const int bh    = task >> 7;            // 128 q-tiles per (b,h): whole block same bh
  const int qbase = (task & 127) * 16;
  const int bb = bh >> 4, h = bh & 15;

  // staging coordinates: 256 threads x 16B cover each 4KB tile exactly
  const int krow = tid >> 3, kc = tid & 7;   // K: 32 rows x 128B
  const int vrow = tid >> 2, vc = tid & 3;   // V: 64 rows x 64B
  const bf16_t* kg = Kb + ((size_t)bh * kT + krow) * kD + kc * 8;
  const bf16_t* vg = Vt + ((size_t)bh * kD + vrow) * kT + vc * 8;

  // A fragments of Q (loop invariant), two K-slices covering d=0..63
  const bf16_t* qrow = Qb + ((size_t)bh * kT + qbase + ln) * kD;
  const v16bf aq0 = load_frag_k32(qrow, lh);
  const v16bf aq1 = load_frag_k32(qrow + 32, lh);

  // loop-invariant per-C-VGPR row constants (row = qbase + 8*lh + r)
  float pm[8], pr[8];
#pragma unroll
  for (int r = 0; r < 8; ++r) {
    int row = qbase + 8 * lh + r;
    pm[r] = mu[(size_t)bh * kT + row];
    pr[r] = rs[(size_t)bh * kT + row];
  }
  const float gate = 1.f / (1.f + __expf(-alp[h]));
  const float gsc  = gate * gmm[h];
  const float gsh  = gate * bet[h];

  float m_i = -INFINITY, l_i = 0.f;       // owned by lanes 0..15 (row == lane)
  v8f acc[4];
#pragma unroll
  for (int j = 0; j < 4; ++j)
#pragma unroll
    for (int i = 0; i < 8; ++i) acc[j][i] = 0.f;

  float*       rawb = raw    + (size_t)bh * kT * kT;
  const float* spb  = s_prev + (size_t)bh * kT * kT;

  // prologue: DMA tiles 0 and 1 (each wave contributes 2 async ops per tile)
  async_load_b128((uint32_t)(uintptr_t)&ldsK[0][krow * 64 + kc * 8], kg);
  async_load_b128((uint32_t)(uintptr_t)&ldsV[0][vrow * 32 + vc * 8], vg);
  async_load_b128((uint32_t)(uintptr_t)&ldsK[1][krow * 64 + kc * 8], kg + (size_t)32 * kD);
  async_load_b128((uint32_t)(uintptr_t)&ldsV[1][vrow * 32 + vc * 8], vg + 32);

  int cur = 0;
  for (int kt = 0; kt < kT; kt += 32) {
    // wait for buf[cur]'s DMA: if a successor tile is in flight keep its 2
    // async ops outstanding, otherwise drain fully.
    if (kt + 32 < kT) asm volatile("s_wait_asynccnt 0x2" ::: "memory");
    else              asm volatile("s_wait_asynccnt 0x0" ::: "memory");
    __syncthreads();  // tile visible to all 8 waves

    // ---- S = Q K^T (16 x 32 tile as two 16x16 C fragments) from LDS K ----
    v8f s0, s1;
#pragma unroll
    for (int i = 0; i < 8; ++i) { s0[i] = 0.f; s1[i] = 0.f; }
    const bf16_t* kr0 = &ldsK[cur][(size_t)ln * 64];
    const bf16_t* kr1 = &ldsK[cur][(size_t)(16 + ln) * 64];
    s0 = wmma_bf16(aq0, load_frag_k32(kr0, lh), s0);
    s0 = wmma_bf16(aq1, load_frag_k32(kr0 + 32, lh), s0);
    s1 = wmma_bf16(aq0, load_frag_k32(kr1, lh), s1);
    s1 = wmma_bf16(aq1, load_frag_k32(kr1 + 32, lh), s1);

    // ---- gated residual + write raw + stage S to LDS ----
#pragma unroll
    for (int r = 0; r < 8; ++r) {
      const int row = qbase + 8 * lh + r;
      const float* sp = spb + (size_t)row * kT + kt;
      float v0 = s0[r] * 0.125f + gsc * ((sp[ln]      - pm[r]) * pr[r]) + gsh;
      float v1 = s1[r] * 0.125f + gsc * ((sp[16 + ln] - pm[r]) * pr[r]) + gsh;
      rawb[(size_t)row * kT + kt + ln]      = v0;
      rawb[(size_t)row * kT + kt + 16 + ln] = v1;
      ldsS[wv][(8 * lh + r) * 32 + ln]      = v0;
      ldsS[wv][(8 * lh + r) * 32 + 16 + ln] = v1;
    }
    wave_lds_fence();

    // ---- online softmax (lanes 0..15: one row each) ----
    if (lane < 16) {
      const float* Sr = &ldsS[wv][lane * 32];
      float tmax = Sr[0];
#pragma unroll
      for (int j = 1; j < 32; ++j) tmax = fmaxf(tmax, Sr[j]);
      const float mnew = fmaxf(m_i, tmax);
      const float corr = __expf(m_i - mnew);
      float sum = 0.f;
#pragma unroll
      for (int j = 0; j < 32; ++j) {
        float p = __expf(Sr[j] - mnew);
        sum += p;
        ldsP[wv][lane * 32 + j] = (bf16_t)p;   // P tile, row-major 16x32
      }
      l_i = l_i * corr + sum;
      m_i = mnew;
      ldsScale[wv][lane] = corr;
    }
    wave_lds_fence();

    // ---- rescale ctx and accumulate P·V from LDS V ----
    const v16bf ap = load_frag_k32(&ldsP[wv][ln * 32], lh); // A frag of P
    float sc[8];
#pragma unroll
    for (int r = 0; r < 8; ++r) sc[r] = ldsScale[wv][8 * lh + r];
#pragma unroll
    for (int j = 0; j < 4; ++j)
#pragma unroll
      for (int r = 0; r < 8; ++r) acc[j][r] *= sc[r];
#pragma unroll
    for (int j = 0; j < 4; ++j) {
      const bf16_t* vr = &ldsV[cur][(size_t)(j * 16 + ln) * 32];
      acc[j] = wmma_bf16(ap, load_frag_k32(vr, lh), acc[j]);
    }

    __syncthreads();  // every wave done reading buf[cur] before reusing it
    if (kt + 64 < kT) {
      async_load_b128((uint32_t)(uintptr_t)&ldsK[cur][krow * 64 + kc * 8],
                      kg + (size_t)(kt + 64) * kD);
      async_load_b128((uint32_t)(uintptr_t)&ldsV[cur][vrow * 32 + vc * 8],
                      vg + (kt + 64));
    }
    cur ^= 1;
  }

  // ---- normalize and write ctx as (B,T,HID) bf16 for the output GEMM ----
  if (lane < 16) ldsL[wv][lane] = l_i;
  wave_lds_fence();
  float inv[8];
#pragma unroll
  for (int r = 0; r < 8; ++r) inv[r] = 1.f / ldsL[wv][8 * lh + r];
#pragma unroll
  for (int j = 0; j < 4; ++j)
#pragma unroll
    for (int r = 0; r < 8; ++r) {
      const int row = qbase + 8 * lh + r;
      const int d   = j * 16 + ln;
      ctx[((size_t)(bb * kT + row)) * kHID + h * kD + d] = (bf16_t)(acc[j][r] * inv[r]);
    }
}

// ---------------------------------------------------------------- launch ----
extern "C" void kernel_launch(void* const* d_in, const int* in_sizes, int n_in,
                              void* d_out, int out_size, void* d_ws, size_t ws_size,
                              hipStream_t stream) {
  const float* x      = (const float*)d_in[0];
  const float* s_prev = (const float*)d_in[1];
  const float* Wq     = (const float*)d_in[2];
  const float* Wk     = (const float*)d_in[3];
  const float* Wv     = (const float*)d_in[4];
  const float* Wo     = (const float*)d_in[5];
  const float* alpha  = (const float*)d_in[6];
  const float* gamma_ = (const float*)d_in[7];
  const float* beta_  = (const float*)d_in[8];
  (void)in_sizes; (void)n_in; (void)out_size; (void)ws_size;

  float* out = (float*)d_out;                      // (B,T,HID) f32
  float* raw = out + (size_t)kBT * kHID;           // (B,H,T,T) f32

  size_t off = 0;
  auto carve = [&](size_t bytes) {
    void* p = (char*)d_ws + off;
    off += (bytes + 255) & ~(size_t)255;
    return p;
  };
  bf16_t* xb  = (bf16_t*)carve((size_t)kBT * kHID * 2);    // x in bf16
  bf16_t* WqT = (bf16_t*)carve((size_t)kHID * kHID * 2);
  bf16_t* WkT = (bf16_t*)carve((size_t)kHID * kHID * 2);
  bf16_t* WvT = (bf16_t*)carve((size_t)kHID * kHID * 2);
  bf16_t* WoT = (bf16_t*)carve((size_t)kHID * kHID * 2);
  bf16_t* Qb  = (bf16_t*)carve((size_t)kBH * kT * kD * 2); // (bh,T,D)
  bf16_t* Kb  = (bf16_t*)carve((size_t)kBH * kT * kD * 2); // (bh,T,D)
  bf16_t* Vt  = (bf16_t*)carve((size_t)kBH * kD * kT * 2); // (bh,D,T)
  bf16_t* ctx = (bf16_t*)carve((size_t)kBT * kHID * 2);    // (B,T,HID)
  float*  mub = (float*)carve((size_t)kBH * kT * 4);
  float*  rsb = (float*)carve((size_t)kBH * kT * 4);

  const int nx = kBT * kHID;
  k_f32_to_bf16<<<nx / 256, 256, 0, stream>>>(x, xb, nx);
  const int nw = kHID * kHID;
  k_transpose_bf16<<<nw / 256, 256, 0, stream>>>(Wq, WqT);
  k_transpose_bf16<<<nw / 256, 256, 0, stream>>>(Wk, WkT);
  k_transpose_bf16<<<nw / 256, 256, 0, stream>>>(Wv, WvT);
  k_transpose_bf16<<<nw / 256, 256, 0, stream>>>(Wo, WoT);
  k_row_stats<<<(kBH * kT) / 8, 256, 0, stream>>>(s_prev, mub, rsb);

  dim3 gg(kBT / 64, kHID / 64);
  k_gemm_bf16<<<gg, 128, 0, stream>>>(xb, WqT, Qb, 0);
  k_gemm_bf16<<<gg, 128, 0, stream>>>(xb, WkT, Kb, 0);
  k_gemm_bf16<<<gg, 128, 0, stream>>>(xb, WvT, Vt, 1);

  k_attn<<<(kBH * (kT / 16)) / 8, 256, 0, stream>>>(Qb, Kb, Vt, s_prev, mub, rsb,
                                                    alpha, gamma_, beta_, raw, ctx);

  k_gemm_bf16<<<gg, 128, 0, stream>>>(ctx, WoT, out, 2);
}
